// AttentionLayer_49125835931773
// MI455X (gfx1250) — matrix operand order repaired and checked by reading, not compiled, not run on hardware
//
#include <hip/hip_runtime.h>
#include <hip/hip_bf16.h>

// ---------------------------------------------------------------------------
// MHA layer for MI455X (gfx1250): all GEMMs on v_wmma_f32_16x16x32_f16.
// B=2, S=2048, D_MODEL=512, H=8, D_K=D_V=64.
// ---------------------------------------------------------------------------

typedef _Float16 h16;
typedef __attribute__((ext_vector_type(16))) _Float16 v16h;
typedef __attribute__((ext_vector_type(8)))  _Float16 v8h;
typedef __attribute__((ext_vector_type(8)))  float    v8f;

#define NB   2
#define NS   2048
#define DM   512
#define NH   8
#define DK   64

union Frag16 { v16h v; v8h h[2]; };

// A-matrix fragment (16x32 f16), row-major source with element stride `stride`.
// ISA layout: lanes 0-15 hold row (lane&15), K chunks {0..7,16..23};
//             lanes 16-31 hold K chunks {8..15,24..31}.
__device__ inline v16h load_a16x32(const h16* __restrict__ base, int stride, int lane) {
  int row = lane & 15;
  int c0  = (lane & 16) ? 8 : 0;
  const h16* p = base + row * stride + c0;
  Frag16 f;
  f.h[0] = *(const v8h*)(p);
  f.h[1] = *(const v8h*)(p + 16);
  return f.v;
}

// B-matrix fragment (32x16 f16) sourced from *transposed* storage Bt[n][k]
// (row = output column n, stride elements). ISA layout: lane (l&15)=column n,
// lanes 0-15 hold K chunks {0..7,16..23}; lanes 16-31 hold K chunks {8..15,24..31}.
__device__ inline v16h load_b32x16(const h16* __restrict__ bt, int stride, int lane) {
  int n  = lane & 15;
  int k  = (lane & 16) ? 8 : 0;
  const h16* p = bt + n * stride + k;
  Frag16 f;
  f.h[0] = *(const v8h*)(p);
  f.h[1] = *(const v8h*)(p + 16);
  return f.v;
}

__device__ inline v8f wmma_f16(v16h a, v16h b, v8f c) {
  // (neg_a, A, neg_b, B, c_mod, C, reuse_a, reuse_b)
  return __builtin_amdgcn_wmma_f32_16x16x32_f16(false, a, false, b, (short)0, c,
                                                false, false);
}

// ---------------------------------------------------------------------------
// Prep: f32 -> f16 convert, and weight transpose+convert (Wt[n][k] = W[k][n]).
// ---------------------------------------------------------------------------
__global__ void k_cvt_f16(const float* __restrict__ src, h16* __restrict__ dst, int n) {
  int i = blockIdx.x * blockDim.x + threadIdx.x;
  if (i < n) dst[i] = (h16)src[i];
}

__global__ void k_transpose512(const float* __restrict__ src, h16* __restrict__ dst) {
  int i = blockIdx.x * blockDim.x + threadIdx.x;   // 512*512 elements
  int n = i >> 9, k = i & 511;
  dst[i] = (h16)src[k * DM + n];                   // dst[n][k] = src[k][n]
}

// ---------------------------------------------------------------------------
// QKV projection: one wave per (16-row m-tile, 16-col n-tile, matrix).
// Q scaled by 1/sqrt(64) and stored head-major [B,H,S,64] (f16).
// K stored head-major [B,H,S,64]; V stored head-TRANSPOSED [B,H,64,S].
// ---------------------------------------------------------------------------
__global__ void k_qkv_proj(const h16* __restrict__ xh,
                           const h16* __restrict__ wqt, const h16* __restrict__ wkt,
                           const h16* __restrict__ wvt,
                           h16* __restrict__ Qh, h16* __restrict__ Kh,
                           h16* __restrict__ Vt) {
  int mt   = blockIdx.x;            // 0..255  (16 rows of B*S=4096)
  int task = blockIdx.y;            // 0..95   (mat * 32 + n-tile)
  int mat  = task >> 5;
  int n0   = (task & 31) << 4;
  int lane = threadIdx.x & 31;

  const h16* wt    = (mat == 0) ? wqt : (mat == 1) ? wkt : wvt;
  const h16* abase = xh + (size_t)mt * 16 * DM;

  v8f c = {0.f, 0.f, 0.f, 0.f, 0.f, 0.f, 0.f, 0.f};
  for (int k0 = 0; k0 < DM; k0 += 32) {
    v16h a = load_a16x32(abase + k0, DM, lane);
    v16h b = load_b32x16(wt + (size_t)n0 * DM + k0, DM, lane);
    c = wmma_f16(a, b, c);
  }

  int N  = lane & 15;
  int m0 = (lane & 16) ? 8 : 0;
  int h  = n0 >> 6;
  int d  = (n0 & 63) + N;
#pragma unroll
  for (int v = 0; v < 8; ++v) {
    int s    = mt * 16 + m0 + v;     // global row in [0, 4096)
    int b_   = s >> 11;
    int srow = s & (NS - 1);
    float val = c[v];
    size_t bh = (size_t)b_ * NH + h;
    if (mat == 0) {
      Qh[(bh * NS + srow) * DK + d] = (h16)(val * 0.125f);
    } else if (mat == 1) {
      Kh[(bh * NS + srow) * DK + d] = (h16)val;
    } else {
      Vt[(bh * DK + d) * NS + srow] = (h16)val;
    }
  }
}

// ---------------------------------------------------------------------------
// Attention: one workgroup (256 thr / 8 waves) per (b, h, 16 query rows).
// LDS: 16x2048 f32 scores (128KB) + 16x2048 f16 attn (64KB) + reductions.
// ---------------------------------------------------------------------------
__global__ void k_attention(const h16* __restrict__ Qh, const h16* __restrict__ Kh,
                            const h16* __restrict__ Vt,
                            float* __restrict__ attn, h16* __restrict__ ctxh) {
  extern __shared__ char smem[];
  float* sc     = (float*)smem;                                  // [16][2048] f32
  h16*   shh    = (h16*)(smem + 16 * NS * 4);                    // [16][2048] f16
  float* red    = (float*)(smem + 16 * NS * 4 + 16 * NS * 2);    // [16][16]
  float* rowmax = red + 256;                                     // [16]
  float* rowsum = rowmax + 16;                                   // [16]

  int bid  = blockIdx.x;          // 0..2047
  int qt   = bid & 127;
  int bh   = bid >> 7;            // b*8 + h
  int q0   = qt * 16;
  int tid  = threadIdx.x;
  int lane = tid & 31;
  int wave = tid >> 5;

  // Q A-fragments (reused across all key tiles): 16x64 -> two 16x32 chunks.
  const h16* qbase = Qh + ((size_t)bh * NS + q0) * DK;
  v16h qa0 = load_a16x32(qbase + 0,  DK, lane);
  v16h qa1 = load_a16x32(qbase + 32, DK, lane);

  // ---- Phase 1: scores = (Q/8) . K^T into LDS --------------------------------
  for (int kt = wave; kt < NS / 16; kt += 8) {
    const h16* kbase = Kh + ((size_t)bh * NS + kt * 16) * DK;
    v8f c = {0.f, 0.f, 0.f, 0.f, 0.f, 0.f, 0.f, 0.f};
    c = wmma_f16(qa0, load_b32x16(kbase + 0,  DK, lane), c);
    c = wmma_f16(qa1, load_b32x16(kbase + 32, DK, lane), c);
    int N  = lane & 15;
    int m0 = (lane & 16) ? 8 : 0;
#pragma unroll
    for (int v = 0; v < 8; ++v)
      sc[(m0 + v) * NS + kt * 16 + N] = c[v];
  }
  __syncthreads();

  // ---- Phase 2: softmax (16 rows x 2048), write attn f32 + f16 copy ----------
  int row = tid >> 4;
  int sub = tid & 15;

  float m = -3.0e38f;
  for (int j = 0; j < NS / 16; ++j)
    m = fmaxf(m, sc[row * NS + sub + 16 * j]);
  red[row * 16 + sub] = m;
  __syncthreads();
  if (sub == 0) {
    float mm = red[row * 16];
#pragma unroll
    for (int i = 1; i < 16; ++i) mm = fmaxf(mm, red[row * 16 + i]);
    rowmax[row] = mm;
  }
  __syncthreads();

  float mrow = rowmax[row];
  float psum = 0.f;
  for (int j = 0; j < NS / 16; ++j) {
    int c = sub + 16 * j;
    float e = __expf(sc[row * NS + c] - mrow);
    sc[row * NS + c] = e;
    psum += e;
  }
  red[row * 16 + sub] = psum;
  __syncthreads();
  if (sub == 0) {
    float ss = 0.f;
#pragma unroll
    for (int i = 0; i < 16; ++i) ss += red[row * 16 + i];
    rowsum[row] = ss;
  }
  __syncthreads();

  float inv = 1.0f / rowsum[row];
  float* arow = attn + ((size_t)bh * NS + q0 + row) * NS;
  for (int j = 0; j < NS / 16; ++j) {
    int c = sub + 16 * j;
    float a = sc[row * NS + c] * inv;
    __builtin_nontemporal_store(a, &arow[c]);   // 268MB stream-out: bypass cache
    shh[row * NS + c] = (h16)a;
  }
  __syncthreads();

  // ---- Phase 3: ctx(16x64) = attn(16x2048) @ V(2048x64), waves 0-3 -----------
  if (wave < 4) {
    int n0 = wave * 16;
    const h16* vtb = Vt + ((size_t)bh * DK + n0) * NS;   // Vt[d][s], contiguous keys
    v8f c = {0.f, 0.f, 0.f, 0.f, 0.f, 0.f, 0.f, 0.f};
    for (int k0 = 0; k0 < NS; k0 += 32) {
      v16h a = load_a16x32(shh + k0, NS, lane);
      v16h b = load_b32x16(vtb + k0, NS, lane);
      c = wmma_f16(a, b, c);
    }
    int N  = lane & 15;
    int m0 = (lane & 16) ? 8 : 0;
    int b_ = bh >> 3, h = bh & 7;
#pragma unroll
    for (int v = 0; v < 8; ++v) {
      size_t srow = (size_t)b_ * NS + q0 + m0 + v;
      ctxh[srow * DM + h * DK + n0 + N] = (h16)c[v];
    }
  }
}

// ---------------------------------------------------------------------------
// Output projection: out = ctx @ Wo + bo, one wave per 16x16 tile.
// ---------------------------------------------------------------------------
__global__ void k_out_proj(const h16* __restrict__ ctxh, const h16* __restrict__ wot,
                           const float* __restrict__ bo, float* __restrict__ out) {
  int mt   = blockIdx.x;          // 0..255
  int n0   = blockIdx.y << 4;     // 0..511
  int lane = threadIdx.x & 31;

  const h16* abase = ctxh + (size_t)mt * 16 * DM;
  v8f c = {0.f, 0.f, 0.f, 0.f, 0.f, 0.f, 0.f, 0.f};
  for (int k0 = 0; k0 < DM; k0 += 32) {
    v16h a = load_a16x32(abase + k0, DM, lane);
    v16h b = load_b32x16(wot + (size_t)n0 * DM + k0, DM, lane);
    c = wmma_f16(a, b, c);
  }
  int N  = lane & 15;
  int m0 = (lane & 16) ? 8 : 0;
  float bias = bo[n0 + N];
#pragma unroll
  for (int v = 0; v < 8; ++v)
    out[((size_t)mt * 16 + m0 + v) * DM + n0 + N] = c[v] + bias;
}

// ---------------------------------------------------------------------------
extern "C" void kernel_launch(void* const* d_in, const int* in_sizes, int n_in,
                              void* d_out, int out_size, void* d_ws, size_t ws_size,
                              hipStream_t stream) {
  const float* x  = (const float*)d_in[0];
  const float* Wq = (const float*)d_in[1];
  const float* Wk = (const float*)d_in[2];
  const float* Wv = (const float*)d_in[3];
  const float* Wo = (const float*)d_in[4];
  const float* bo = (const float*)d_in[5];

  // Workspace carve-up (~23 MB total, all 256B-aligned chunks).
  char* ws = (char*)d_ws;
  h16* xh   = (h16*)ws;  ws += (size_t)NB * NS * DM * 2;        // 4 MB
  h16* wqt  = (h16*)ws;  ws += (size_t)DM * DM * 2;             // 512 KB
  h16* wkt  = (h16*)ws;  ws += (size_t)DM * DM * 2;
  h16* wvt  = (h16*)ws;  ws += (size_t)DM * DM * 2;
  h16* wot  = (h16*)ws;  ws += (size_t)DM * DM * 2;
  h16* Qh   = (h16*)ws;  ws += (size_t)NB * NH * NS * DK * 2;   // 4 MB
  h16* Kh   = (h16*)ws;  ws += (size_t)NB * NH * NS * DK * 2;
  h16* Vt   = (h16*)ws;  ws += (size_t)NB * NH * NS * DK * 2;
  h16* ctxh = (h16*)ws;  ws += (size_t)NB * NS * DM * 2;        // 4 MB

  float* out  = (float*)d_out;                         // [B,S,512]
  float* attn = out + (size_t)NB * NS * DM;            // [B,H,S,S]

  // Stage 0: conversions / transposes.
  int nx = NB * NS * DM;
  k_cvt_f16<<<(nx + 255) / 256, 256, 0, stream>>>(x, xh, nx);
  k_transpose512<<<(DM * DM) / 256, 256, 0, stream>>>(Wq, wqt);
  k_transpose512<<<(DM * DM) / 256, 256, 0, stream>>>(Wk, wkt);
  k_transpose512<<<(DM * DM) / 256, 256, 0, stream>>>(Wv, wvt);
  k_transpose512<<<(DM * DM) / 256, 256, 0, stream>>>(Wo, wot);

  // Stage 1: QKV projections (WMMA).
  dim3 gQKV(NB * NS / 16, 96);                 // 256 m-tiles x (3 mats * 32 n-tiles)
  k_qkv_proj<<<gQKV, 32, 0, stream>>>(xh, wqt, wkt, wvt, Qh, Kh, Vt);

  // Stage 2: attention (WMMA scores -> softmax in LDS -> WMMA attn@V).
  size_t lds = (size_t)16 * NS * 4 + (size_t)16 * NS * 2 + (256 + 32) * 4;  // ~194 KB
  k_attention<<<NB * NH * (NS / 16), 256, lds, stream>>>(Qh, Kh, Vt, attn, ctxh);

  // Stage 3: output projection + bias (WMMA).
  dim3 gO(NB * NS / 16, DM / 16);
  k_out_proj<<<gO, 32, 0, stream>>>(ctxh, wot, bo, out);
}
